// DGCNN_8151847928116
// MI455X (gfx1250) — matrix operand order repaired and checked by reading, not compiled, not run on hardware
//
#include <hip/hip_runtime.h>
#include <hip/hip_bf16.h>

typedef __attribute__((ext_vector_type(16))) _Float16 v16h;
typedef __attribute__((ext_vector_type(8)))  _Float16 v8h;
typedef __attribute__((ext_vector_type(8)))  float    v8f;

#define KNN 16

__device__ __forceinline__ v8h sub8(v8h a, v8h b) {  // a - b, packed f16
  v8h r;
#pragma unroll
  for (int e = 0; e < 8; ++e) r[e] = a[e] - b[e];
  return r;
}

// Load 8 contiguous halves of the concatenated edge feature [x_i | x_j - x_i]
// starting at column k. Requires CREAL % 8 == 0 so a chunk never straddles the
// concat boundary (true for layers 2 and 3).
template<int CREAL>
__device__ __forceinline__ v8h load_cat8(const _Float16* __restrict__ xi,
                                         const _Float16* __restrict__ xj, int k) {
  if (k < CREAL) {
    return *(const v8h*)(xi + k);
  } else {
    const int kk = k - CREAL;
    v8h a = *(const v8h*)(xi + kk);
    v8h b = *(const v8h*)(xj + kk);
    return sub8(b, a);
  }
}

// Build the full set of A fragments (ISA 16-bit A layout: lane r holds row r,
// two contiguous 8-half chunks at kb+klo and kb+16+klo) for one node's
// 16 x K2 edge-feature tile (rows = the node's 16 kNN neighbors).
template<int CREAL, int K2>
__device__ __forceinline__ void build_afrag(const _Float16* __restrict__ xin,
                                            const long long* __restrict__ src,
                                            int node, int r, int klo,
                                            v16h* __restrict__ afrag) {
  const long long j = src[(size_t)node * KNN + r];
  const _Float16* xi = xin + (size_t)node * CREAL;
  const _Float16* xj = xin + (size_t)j    * CREAL;

  if constexpr ((CREAL % 8) == 0) {
#pragma unroll
    for (int ks = 0; ks < K2 / 32; ++ks) {
      const int kb = ks * 32;
      v8h c0 = load_cat8<CREAL>(xi, xj, kb + klo);
      v8h c1 = load_cat8<CREAL>(xi, xj, kb + 16 + klo);
      v16h a;
#pragma unroll
      for (int e = 0; e < 8; ++e) { a[e] = c0[e]; a[8 + e] = c1[e]; }
      afrag[ks] = a;
    }
  } else {                                 // layer 1: C=3, scalar gather + pad
#pragma unroll
    for (int ks = 0; ks < K2 / 32; ++ks) {
      v16h a;
#pragma unroll
      for (int e = 0; e < 16; ++e) {
        const int k = ks * 32 + ((e < 8) ? 0 : 16) + klo + (e & 7);
        _Float16 v = (_Float16)0.0f;
        if (k < CREAL)        v = xi[k];
        else if (k < 2*CREAL) v = (_Float16)((float)xj[k - CREAL] - (float)xi[k - CREAL]);
        a[e] = v;
      }
      afrag[ks] = a;
    }
  }
}

// Epilogue: bias + relu, max over the 16 rows (8 accum VGPRs in-lane, then
// cross-half-lane via shfl_xor 16), store one f16 per column from lanes 0-15.
__device__ __forceinline__ void edge_epilogue(v8f acc, float bias, int hi, int r,
                                              _Float16* __restrict__ xout,
                                              size_t node_base, int ncol) {
  float best = 0.0f;                       // relu >= 0; every node has 16 edges
#pragma unroll
  for (int d = 0; d < 8; ++d) best = fmaxf(best, fmaxf(acc[d] + bias, 0.0f));
  best = fmaxf(best, __shfl_xor(best, 16, 32));
  if (hi == 0) xout[node_base + ncol] = (_Float16)best;
}

// Fused EdgeConv. One wave processes TWO nodes per iteration so that each B
// fragment read from LDS feeds two independent WMMA chains (halves LDS
// traffic per FLOP and hides WMMA accumulation latency).
template<int CREAL, int K2, int NTILE, int COUT>
__global__ __launch_bounds__(256) void edgeconv_kernel(
    const _Float16* __restrict__ xin, const long long* __restrict__ src,
    const float* __restrict__ W, const float* __restrict__ bvec,
    _Float16* __restrict__ xout, int n_nodes)
{
  __shared__ _Float16 Wt[NTILE * K2];      // transposed weights: Wt[n][k]
  const int ncb = blockIdx.y * NTILE;

  for (int idx = threadIdx.x; idx < NTILE * K2; idx += blockDim.x) {
    const int nn = idx / K2;
    const int kk = idx - nn * K2;
    float w = (kk < 2 * CREAL) ? W[(size_t)kk * COUT + (ncb + nn)] : 0.0f;
    Wt[idx] = (_Float16)w;
  }
  __syncthreads();

  const int lane = threadIdx.x & 31;
  const int r    = lane & 15;              // row within the 16x16 tile
  const int hi   = lane >> 4;              // lane half (0/1)
  const int klo  = hi ? 8 : 0;             // ISA A-layout K offset, upper lanes
  const int nwaves  = blockDim.x >> 5;
  const int wave    = blockIdx.x * nwaves + (threadIdx.x >> 5);
  const int wstride = gridDim.x * nwaves;

  for (int n0 = wave * 2; n0 < n_nodes; n0 += wstride * 2) {
    const int n1 = (n0 + 1 < n_nodes) ? (n0 + 1) : n0;   // clamp tail (benign dup)

    v16h afA[K2 / 32], afB[K2 / 32];
    build_afrag<CREAL, K2>(xin, src, n0, r, klo, afA);
    build_afrag<CREAL, K2>(xin, src, n1, r, klo, afB);

#pragma unroll
    for (int nt = 0; nt < NTILE / 16; ++nt) {
      v8f accA = {}, accB = {};
      // B fragment: lane's column = nt*16 + r; two 8-half K chunks per lane at
      // 8*hi and 16+8*hi (same K order as A). Two ds_load_b128 per fragment,
      // each shared by two WMMAs.
      const _Float16* wt = &Wt[(size_t)(nt * 16 + r) * K2 + (hi ? 8 : 0)];
#pragma unroll
      for (int ks = 0; ks < K2 / 32; ++ks) {
        const v8h b0 = *(const v8h*)(wt + ks * 32);
        const v8h b1 = *(const v8h*)(wt + ks * 32 + 16);
        v16h bf = __builtin_shufflevector(b0, b1, 0, 1, 2, 3, 4, 5, 6, 7, 8, 9, 10, 11, 12, 13, 14, 15);
        accA = __builtin_amdgcn_wmma_f32_16x16x32_f16(false, afA[ks], false, bf,
                                                      (short)0, accA, false, false);
        accB = __builtin_amdgcn_wmma_f32_16x16x32_f16(false, afB[ks], false, bf,
                                                      (short)0, accB, false, false);
      }
      const int   ncol = ncb + nt * 16 + r;
      const float bias = bvec[ncol];
      edge_epilogue(accA, bias, hi, r, xout, (size_t)n0 * COUT, ncol);
      edge_epilogue(accB, bias, hi, r, xout, (size_t)n1 * COUT, ncol);
    }
  }
}

// Plain WMMA GEMM with fused bias+relu: out = relu(A @ W + b), A f16 [n_rows][K].
template<int K, int NTILE, int COUT>
__global__ __launch_bounds__(256) void gemm_relu_kernel(
    const _Float16* __restrict__ A, const float* __restrict__ W,
    const float* __restrict__ bvec, _Float16* __restrict__ out, int n_rows)
{
  __shared__ _Float16 Wt[NTILE * K];
  const int ncb = blockIdx.y * NTILE;
  for (int idx = threadIdx.x; idx < NTILE * K; idx += blockDim.x) {
    const int nn = idx / K;
    const int kk = idx - nn * K;
    Wt[idx] = (_Float16)W[(size_t)kk * COUT + (ncb + nn)];
  }
  __syncthreads();

  const int lane = threadIdx.x & 31;
  const int r    = lane & 15;
  const int hi   = lane >> 4;
  const int klo  = hi ? 8 : 0;
  const int nwaves  = blockDim.x >> 5;
  const int wave    = blockIdx.x * nwaves + (threadIdx.x >> 5);
  const int wstride = gridDim.x * nwaves;
  const int mtiles  = (n_rows + 15) >> 4;

  for (int mt = wave; mt < mtiles; mt += wstride) {
    int row = mt * 16 + r;
    if (row >= n_rows) row = n_rows - 1;
    const _Float16* xr = A + (size_t)row * K;

    v16h afrag[K / 32];
#pragma unroll
    for (int ks = 0; ks < K / 32; ++ks) {
      v8h c0 = *(const v8h*)(xr + ks * 32 + klo);
      v8h c1 = *(const v8h*)(xr + ks * 32 + 16 + klo);
      v16h a;
#pragma unroll
      for (int e = 0; e < 8; ++e) { a[e] = c0[e]; a[8 + e] = c1[e]; }
      afrag[ks] = a;
    }

#pragma unroll
    for (int nt = 0; nt < NTILE / 16; ++nt) {
      v8f acc = {};
      const _Float16* wt = &Wt[(size_t)(nt * 16 + r) * K + (hi ? 8 : 0)];
#pragma unroll
      for (int ks = 0; ks < K / 32; ++ks) {
        const v8h b0 = *(const v8h*)(wt + ks * 32);
        const v8h b1 = *(const v8h*)(wt + ks * 32 + 16);
        v16h bf = __builtin_shufflevector(b0, b1, 0, 1, 2, 3, 4, 5, 6, 7, 8, 9, 10, 11, 12, 13, 14, 15);
        acc = __builtin_amdgcn_wmma_f32_16x16x32_f16(false, afrag[ks], false, bf,
                                                     (short)0, acc, false, false);
      }
      const int   ncol  = ncb + nt * 16 + r;
      const float bias  = bvec[ncol];
      const int   rbase = mt * 16 + (hi ? 8 : 0);   // ISA C/D layout rows
#pragma unroll
      for (int d = 0; d < 8; ++d) {
        const int orow = rbase + d;
        if (orow < n_rows) {
          const float v = fmaxf(acc[d] + bias, 0.0f);
          out[(size_t)orow * COUT + ncol] = (_Float16)v;
        }
      }
    }
  }
}

__global__ void convert_f32_to_f16(const float* __restrict__ in,
                                   _Float16* __restrict__ outp, int count) {
  const int i = blockIdx.x * blockDim.x + threadIdx.x;
  if (i < count) outp[i] = (_Float16)in[i];
}

// out[i][c] = x[i][c] + bl2[c] + sum_k h[i][k] * Wl2[k][c]   (c = 0..2)
__global__ void final_kernel(const float* __restrict__ x, const _Float16* __restrict__ h,
                             const float* __restrict__ Wl2, const float* __restrict__ bl2,
                             float* __restrict__ out, int n) {
  const int i = blockIdx.x * blockDim.x + threadIdx.x;
  if (i >= n) return;
  const _Float16* hr = h + (size_t)i * 256;
  float a0 = bl2[0], a1 = bl2[1], a2 = bl2[2];
#pragma unroll 4
  for (int k = 0; k < 256; ++k) {
    const float hv = (float)hr[k];
    a0 += hv * Wl2[k * 3 + 0];
    a1 += hv * Wl2[k * 3 + 1];
    a2 += hv * Wl2[k * 3 + 2];
  }
  out[i * 3 + 0] = x[i * 3 + 0] + a0;
  out[i * 3 + 1] = x[i * 3 + 1] + a1;
  out[i * 3 + 2] = x[i * 3 + 2] + a2;
}

extern "C" void kernel_launch(void* const* d_in, const int* in_sizes, int n_in,
                              void* d_out, int out_size, void* d_ws, size_t ws_size,
                              hipStream_t stream) {
  const float*     x   = (const float*)d_in[0];
  const long long* ei  = (const long long*)d_in[1];   // [2][E] int64
  const float* W1 = (const float*)d_in[2];  const float* b1  = (const float*)d_in[3];
  const float* W2 = (const float*)d_in[4];  const float* b2  = (const float*)d_in[5];
  const float* W3 = (const float*)d_in[6];  const float* b3  = (const float*)d_in[7];
  const float* Wl1= (const float*)d_in[8];  const float* bl1 = (const float*)d_in[9];
  const float* Wl2= (const float*)d_in[10]; const float* bl2 = (const float*)d_in[11];
  float* out = (float*)d_out;

  const int n = in_sizes[0] / 3;            // 30000 nodes
  const long long* src = ei;                // row 0 of edge_index = source nodes
  (void)n_in; (void)out_size; (void)ws_size;

  char* ws = (char*)d_ws;
  auto alloc = [&](size_t bytes) -> char* {
    char* p = ws;
    ws += (bytes + 255) & ~(size_t)255;
    return p;
  };
  _Float16* x0h = (_Float16*)alloc((size_t)n * 3   * sizeof(_Float16));
  _Float16* x1h = (_Float16*)alloc((size_t)n * 64  * sizeof(_Float16));
  _Float16* x2h = (_Float16*)alloc((size_t)n * 128 * sizeof(_Float16));
  _Float16* x3h = (_Float16*)alloc((size_t)n * 512 * sizeof(_Float16));
  _Float16* hh  = (_Float16*)alloc((size_t)n * 256 * sizeof(_Float16));

  convert_f32_to_f16<<<dim3((n * 3 + 255) / 256), dim3(256), 0, stream>>>(x, x0h, n * 3);

  // Layer 1: C=3  -> 64   (K2 = 32 padded, scalar gather path)
  edgeconv_kernel<3, 32, 64, 64>
      <<<dim3(256, 1), dim3(256), 0, stream>>>(x0h, src, W1, b1, x1h, n);
  // Layer 2: C=64 -> 128  (K2 = 128, LDS weights 32KB)
  edgeconv_kernel<64, 128, 128, 128>
      <<<dim3(256, 1), dim3(256), 0, stream>>>(x1h, src, W2, b2, x2h, n);
  // Layer 3: C=128 -> 512 (K2 = 256, LDS 64KB per 128-column slice, grid.y = 4)
  edgeconv_kernel<128, 256, 128, 512>
      <<<dim3(256, 4), dim3(256), 0, stream>>>(x2h, src, W3, b3, x3h, n);
  // MLP layer 1: h = relu(x3 @ Wl1 + bl1), K = 512, 64-column slices (LDS 64KB)
  gemm_relu_kernel<512, 64, 256>
      <<<dim3(128, 4), dim3(256), 0, stream>>>(x3h, Wl1, bl1, hh, n);
  // MLP layer 2 + residual (tiny: 46 MFLOP, VALU)
  final_kernel<<<dim3((n + 255) / 256), dim3(256), 0, stream>>>(x, hh, Wl2, bl2, out, n);
}
